// CausalSelfAttention_37598143709263
// MI455X (gfx1250) — matrix operand order repaired and checked by reading, not compiled, not run on hardware
//
#include <hip/hip_runtime.h>
#include <hip/hip_bf16.h>

typedef __bf16 bf16;
typedef __bf16 v16bf __attribute__((ext_vector_type(16)));
typedef __bf16 v8bf  __attribute__((ext_vector_type(8)));
typedef float  v8f   __attribute__((ext_vector_type(8)));

#define B_    4
#define T_    1705
#define C_    768
#define H_    12
#define HD_   64
#define TP_   1728              // T padded to multiple of 32 (k-tile safety)
#define M_    (B_ * T_)         // 6820
#define NQKV_ 2304
#define NQB_  ((T_ + 15) / 16)  // 107 q-blocks per (b,h)

__device__ __forceinline__ bf16 f2bf(float f) {
  unsigned u = __float_as_uint(f);
  u += 0x7FFFu + ((u >> 16) & 1u);               // round-to-nearest-even
  unsigned short s = (unsigned short)(u >> 16);
  return __builtin_bit_cast(bf16, s);
}

// CDNA5 async copy: global -> LDS, 16B per lane, tracked by ASYNCcnt.
__device__ __forceinline__ void async_b128(const bf16* gp, bf16* lp) {
  unsigned lofs = (unsigned)(uintptr_t)lp;  // low 32 bits = LDS offset
  asm volatile("global_load_async_to_lds_b128 %0, %1, off"
               :: "v"(lofs), "v"(gp) : "memory");
}

// A-fragment (16x32 bf16): lane holds row ln; K pattern {0..7,16..23} (lanes 0-15)
// or {8..15,24..31} (lanes 16-31)  -> two 16B loads at +hi*8 and +hi*8+16 elems.
__device__ __forceinline__ v16bf frag_a(const bf16* rowPtr, int hi) {
  v8bf lo = *(const v8bf*)(rowPtr + hi * 8);
  v8bf hi8 = *(const v8bf*)(rowPtr + hi * 8 + 16);
  v16bf r;
#pragma unroll
  for (int i = 0; i < 8; ++i) { r[i] = lo[i]; r[i + 8] = hi8[i]; }
  return r;
}

// B-fragment (32x16 bf16, stored as B^T rows [n][k]): lane holds col ln;
// K pattern {0..7,16..23} (lanes 0-15) / {8..15,24..31} (lanes 16-31).
__device__ __forceinline__ v16bf frag_b(const bf16* rowPtr, int hi) {
  v8bf lo = *(const v8bf*)(rowPtr + hi * 8);
  v8bf hi8 = *(const v8bf*)(rowPtr + hi * 8 + 16);
  v16bf r;
#pragma unroll
  for (int i = 0; i < 8; ++i) { r[i] = lo[i]; r[i + 8] = hi8[i]; }
  return r;
}

__device__ __forceinline__ v8f wmma_bf16(v16bf a, v16bf b, v8f c) {
  return __builtin_amdgcn_wmma_f32_16x16x32_bf16(false, a, false, b, (short)0, c,
                                                 false, false);
}

// ---------------------------------------------------------------- converters
__global__ void cvt_f32_bf16(const float* __restrict__ in, bf16* __restrict__ out,
                             int n) {
  int i = blockIdx.x * blockDim.x + threadIdx.x;
  if (i < n) out[i] = f2bf(in[i]);
}

// in: [K][N] row-major fp32 -> out: [N][K] bf16 (transposed)
__global__ void cvt_transpose_bf16(const float* __restrict__ in,
                                   bf16* __restrict__ out, int K, int N) {
  int i = blockIdx.x * blockDim.x + threadIdx.x;
  if (i < K * N) {
    int n = i / K;
    int k = i - n * K;
    out[i] = f2bf(in[(size_t)k * N + n]);
  }
}

// ----------------------------------------------------------------- WMMA GEMM
// C[M,N] = A[M,K] * Bt[N,K]^T ; workgroup tile 64x128, 8 waves (each 16x64),
// K-step 32, bf16 in, f32 accumulate. LDS tiles are double-buffered and filled
// with CDNA5 async global->LDS copies so the next tile's HBM/L2 traffic
// overlaps the current tile's WMMAs. All fragments for one K-step are loaded
// before the 4 WMMAs so DS latency overlaps matrix issue.
// mode 0: QKV epilogue (scatter into Q[bh][TP][64] *0.125, K[bh][TP][64],
//         Vt[bh][64][TP] as bf16);  mode 1: plain fp32 store to outF.
__global__ __launch_bounds__(256) void gemm_bf16(
    const bf16* __restrict__ A, const bf16* __restrict__ Bt, int Mrows, int Ncols,
    int Kdim, int mode, float* __restrict__ outF, bf16* __restrict__ qO,
    bf16* __restrict__ kO, bf16* __restrict__ vO) {
  __shared__ __align__(16) bf16 As[2][64 * 40];    // stride 40 elems = 80B
  __shared__ __align__(16) bf16 Bs[2][128 * 40];

  const int tid = threadIdx.x;
  const int mBase = blockIdx.x * 64;
  const int nBase = blockIdx.y * 128;
  const int w = tid >> 5, lane = tid & 31, ln = lane & 15, hi = lane >> 4;
  const int wm = w & 3, wn = w >> 2;

  // Per-thread staging coordinates (1 A seg + 2 B segs, 16B each).
  const int rowA = tid >> 2, segA = tid & 3;
  int gmA = mBase + rowA;
  if (gmA >= Mrows) gmA = Mrows - 1;  // clamp: keeps async issue non-divergent
  const bf16* gA = A + (size_t)gmA * Kdim + segA * 8;
  bf16* lA = &As[0][rowA * 40 + segA * 8];

  const int rowB0 = tid >> 2, segB0 = tid & 3;            // rows 0..63
  const int rowB1 = (tid + 256) >> 2, segB1 = tid & 3;    // rows 64..127
  const bf16* gB0 = Bt + (size_t)(nBase + rowB0) * Kdim + segB0 * 8;
  const bf16* gB1 = Bt + (size_t)(nBase + rowB1) * Kdim + segB1 * 8;
  bf16* lB0 = &Bs[0][rowB0 * 40 + segB0 * 8];
  bf16* lB1 = &Bs[0][rowB1 * 40 + segB1 * 8];
  const int bufStrideA = 64 * 40, bufStrideB = 128 * 40;

  v8f acc[4] = {{}, {}, {}, {}};
  const int nk = Kdim >> 5;

  // Prologue: async-stage tile 0 into buffer 0.
  async_b128(gA, lA);
  async_b128(gB0, lB0);
  async_b128(gB1, lB1);

  for (int kt = 0; kt < nk; ++kt) {
    __syncthreads();  // everyone done reading buf[(kt+1)&1] (tile kt-1)
    if (kt + 1 < nk) {
      int nb = (kt + 1) & 1;
      int koff = (kt + 1) << 5;
      async_b128(gA + koff, lA + nb * bufStrideA);
      async_b128(gB0 + koff, lB0 + nb * bufStrideB);
      async_b128(gB1 + koff, lB1 + nb * bufStrideB);
      // 3 ops now in flight for tile kt+1; waiting to <=3 means the (older,
      // in-order) 3 ops of tile kt have retired into LDS.
      asm volatile("s_wait_asynccnt 0x3" ::: "memory");
    } else {
      asm volatile("s_wait_asynccnt 0x0" ::: "memory");
    }
    __syncthreads();  // all waves' tile-kt data visible

    const bf16* as = &As[kt & 1][0];
    const bf16* bs = &Bs[kt & 1][0];
    v16bf aF = frag_a(&as[(wm * 16 + ln) * 40], hi);
    v16bf bF[4];
#pragma unroll
    for (int j = 0; j < 4; ++j)
      bF[j] = frag_b(&bs[(wn * 64 + j * 16 + ln) * 40], hi);
#pragma unroll
    for (int j = 0; j < 4; ++j) acc[j] = wmma_bf16(aF, bF[j], acc[j]);
  }

  if (mode == 1) {
#pragma unroll
    for (int j = 0; j < 4; ++j)
#pragma unroll
      for (int r = 0; r < 8; ++r) {
        int gm = mBase + wm * 16 + r + hi * 8;
        int gn = nBase + wn * 64 + j * 16 + ln;
        if (gm < Mrows) outF[(size_t)gm * Ncols + gn] = acc[j][r];
      }
  } else {
#pragma unroll
    for (int j = 0; j < 4; ++j)
#pragma unroll
      for (int r = 0; r < 8; ++r) {
        int gm = mBase + wm * 16 + r + hi * 8;
        int gn = nBase + wn * 64 + j * 16 + ln;
        if (gm < Mrows) {
          int b = gm / T_;
          int t = gm - b * T_;
          int sec = gn / C_;
          int c = gn - sec * C_;
          int hh = c >> 6;
          int d = c & 63;
          size_t bh = (size_t)(b * H_ + hh);
          float v = acc[j][r];
          if (sec == 0)
            qO[(bh * TP_ + t) * HD_ + d] = f2bf(v * 0.125f);  // 1/sqrt(64)
          else if (sec == 1)
            kO[(bh * TP_ + t) * HD_ + d] = f2bf(v);
          else
            vO[(bh * HD_ + d) * TP_ + t] = f2bf(v);  // V stored transposed
        }
      }
  }
}

// ---------------------------------------------------- flash attention (wave)
// One wave handles one 16-row q-block of one (b,h); loops k in chunks of 32.
__global__ __launch_bounds__(128) void attn_kernel(
    const bf16* __restrict__ Q, const bf16* __restrict__ K,
    const bf16* __restrict__ Vt, const unsigned char* __restrict__ mask,
    bf16* __restrict__ Y) {
  __shared__ __align__(16) bf16 Pb[4][16 * 40];  // per-wave P-tile scratch

  const int w = threadIdx.x >> 5, lane = threadIdx.x & 31;
  const int ln = lane & 15, hi = lane >> 4;
  const int gid = blockIdx.x * 4 + w;
  if (gid >= B_ * H_ * NQB_) return;  // no block barriers in this kernel

  const int qblk = gid % NQB_;
  const int bh = gid / NQB_;
  const int qt0 = qblk * 16;

  const bf16* qp = Q + ((size_t)bh * TP_ + qt0) * HD_;
  const bf16* kp = K + (size_t)bh * TP_ * HD_;
  const bf16* vp = Vt + (size_t)bh * HD_ * TP_;
  bf16* pb = &Pb[w][0];

  v16bf aQ[2];
  aQ[0] = frag_a(qp + ln * HD_, hi);
  aQ[1] = frag_a(qp + ln * HD_ + 32, hi);

  float mrow[8], lrow[8];
#pragma unroll
  for (int r = 0; r < 8; ++r) { mrow[r] = -3.0e38f; lrow[r] = 0.f; }
  v8f o[4] = {{}, {}, {}, {}};

  for (int kc = 0; kc < TP_; kc += 32) {
    // Batch all 4 K-fragment loads (global b128 pairs) before the 4 WMMAs.
    v16bf bK[4];
#pragma unroll
    for (int half = 0; half < 2; ++half) {
      bK[half * 2 + 0] = frag_b(kp + (size_t)(kc + ln) * HD_ + half * 32, hi);
      bK[half * 2 + 1] =
          frag_b(kp + (size_t)(kc + 16 + ln) * HD_ + half * 32, hi);
    }
    v8f s0 = {}, s1 = {};
#pragma unroll
    for (int half = 0; half < 2; ++half) {  // d-window 0..31 / 32..63
      s0 = wmma_bf16(aQ[half], bK[half * 2 + 0], s0);
      s1 = wmma_bf16(aQ[half], bK[half * 2 + 1], s1);
    }

#pragma unroll
    for (int r = 0; r < 8; ++r) {
      int q = qt0 + r + hi * 8;
      int c0 = kc + ln, c1 = kc + ln + 16;
      float v0 = s0[r], v1 = s1[r];
      unsigned char m0 = 0, m1 = 0;
      if (q < T_) {
        const unsigned char* mp = mask + (size_t)q * T_;
        if (c0 < T_) m0 = mp[c0];
        if (c1 < T_) m1 = mp[c1];
      }
      v0 = m0 ? v0 : -1e9f;
      v1 = m1 ? v1 : -1e9f;
      float mx = fmaxf(v0, v1);
      mx = fmaxf(mx, __shfl_xor(mx, 1, 16));
      mx = fmaxf(mx, __shfl_xor(mx, 2, 16));
      mx = fmaxf(mx, __shfl_xor(mx, 4, 16));
      mx = fmaxf(mx, __shfl_xor(mx, 8, 16));
      float mnew = fmaxf(mrow[r], mx);
      float corr = __expf(mrow[r] - mnew);
      mrow[r] = mnew;
      float p0 = __expf(v0 - mnew);
      float p1 = __expf(v1 - mnew);
      float ps = p0 + p1;
      ps += __shfl_xor(ps, 1, 16);
      ps += __shfl_xor(ps, 2, 16);
      ps += __shfl_xor(ps, 4, 16);
      ps += __shfl_xor(ps, 8, 16);
      lrow[r] = lrow[r] * corr + ps;
#pragma unroll
      for (int t = 0; t < 4; ++t) o[t][r] *= corr;
      pb[(r + hi * 8) * 40 + ln] = f2bf(p0);
      pb[(r + hi * 8) * 40 + ln + 16] = f2bf(p1);
    }

    asm volatile("s_wait_dscnt 0x0" ::: "memory");  // P-tile visible to wave

    v16bf aP = frag_a(pb + ln * 40, hi);
    v16bf bV[4];
#pragma unroll
    for (int t = 0; t < 4; ++t)  // V^T rows are contiguous along t
      bV[t] = frag_b(vp + (size_t)(t * 16 + ln) * TP_ + kc, hi);
#pragma unroll
    for (int t = 0; t < 4; ++t) o[t] = wmma_bf16(aP, bV[t], o[t]);
  }

  const int b = bh / H_, h = bh % H_;
#pragma unroll
  for (int r = 0; r < 8; ++r) {
    int q = qt0 + r + hi * 8;
    if (q < T_) {
      float inv = 1.f / lrow[r];
#pragma unroll
      for (int t = 0; t < 4; ++t)
        Y[((size_t)(b * T_ + q)) * C_ + h * HD_ + t * 16 + ln] =
            f2bf(o[t][r] * inv);
    }
  }
}

// ------------------------------------------------------------------- launch
extern "C" void kernel_launch(void* const* d_in, const int* in_sizes, int n_in,
                              void* d_out, int out_size, void* d_ws,
                              size_t ws_size, hipStream_t stream) {
  const float* x = (const float*)d_in[0];
  const float* Wa = (const float*)d_in[1];
  const float* Wp = (const float*)d_in[2];
  const unsigned char* mask = (const unsigned char*)d_in[3];
  float* out = (float*)d_out;

  bf16* ws = (bf16*)d_ws;
  size_t off = 0;
  bf16* xbf = ws + off;  off += (size_t)M_ * C_;
  bf16* waT = ws + off;  off += (size_t)NQKV_ * C_;
  bf16* wpT = ws + off;  off += (size_t)C_ * C_;
  bf16* qb  = ws + off;  off += (size_t)B_ * H_ * TP_ * HD_;
  bf16* kb  = ws + off;  off += (size_t)B_ * H_ * TP_ * HD_;
  bf16* vt  = ws + off;  off += (size_t)B_ * H_ * TP_ * HD_;
  bf16* yb  = ws + off;  off += (size_t)M_ * C_;

  int nx = M_ * C_;
  cvt_f32_bf16<<<(nx + 255) / 256, 256, 0, stream>>>(x, xbf, nx);
  cvt_transpose_bf16<<<(NQKV_ * C_ + 255) / 256, 256, 0, stream>>>(Wa, waT, C_,
                                                                   NQKV_);
  cvt_transpose_bf16<<<(C_ * C_ + 255) / 256, 256, 0, stream>>>(Wp, wpT, C_, C_);

  dim3 g1((M_ + 63) / 64, NQKV_ / 128);
  gemm_bf16<<<g1, 256, 0, stream>>>(xbf, waT, M_, NQKV_, C_, 0, nullptr, qb, kb,
                                    vt);

  int waves = B_ * H_ * NQB_;
  attn_kernel<<<(waves + 3) / 4, 128, 0, stream>>>(qb, kb, vt, mask, yb);

  dim3 g2((M_ + 63) / 64, C_ / 128);
  gemm_bf16<<<g2, 256, 0, stream>>>(yb, wpT, M_, C_, C_, 1, out, nullptr,
                                    nullptr, nullptr);
}
